// CTViT_12463995093711
// MI455X (gfx1250) — matrix operand order repaired and checked by reading, not compiled, not run on hardware
//
#include <hip/hip_runtime.h>
#include <hip/hip_bf16.h>
#include <math.h>

typedef __attribute__((ext_vector_type(16))) _Float16 v16h;
typedef __attribute__((ext_vector_type(8)))  float    v8f;

// ---------------- problem constants ----------------
#define TT_    16          // Tt
#define BT_    64          // B*Tt
#define MTOK_  16384       // BT*N tokens
#define DIM_   512
#define HEADS_ 8
#define FFI_   1365
#define FFP_   1408        // FFI padded to 64-multiple
#define DEPTH_ 4
#define NPAIR_ 65536       // N*N

// ---------------- WMMA fragment helpers (CDNA5 wave32 layouts) -------------
// A (16x32 f16) from [m][k] row-major: lane L=(half<<4)|r holds row r;
// elem t -> k = 16*half + t  (one contiguous 16-half chunk)
__device__ __forceinline__ v16h load_a_frag(const _Float16* base, int ld) {
  const int lane = threadIdx.x & 31;
  const int r = lane & 15, half = lane >> 4;
  v16h a;
#pragma unroll
  for (int t = 0; t < 16; ++t) {
    const int k = 16 * half + t;
    a[t] = base[r * ld + k];
  }
  return a;
}
// B (32x16) when memory is [n][k] row-major (k fastest): lane holds col c;
// elem t -> k = 16*half + t  (one contiguous 16-half chunk)
__device__ __forceinline__ v16h load_b_frag_nk(const _Float16* base, int ld) {
  const int lane = threadIdx.x & 31;
  const int c = lane & 15, half = lane >> 4;
  v16h b;
#pragma unroll
  for (int t = 0; t < 16; ++t) b[t] = base[c * ld + 16 * half + t];
  return b;
}

#define WMMA32(a, b, c) \
  __builtin_amdgcn_wmma_f32_16x16x32_f16(false, (a), false, (b), (short)0, (c), false, false)

// ---------------- generic WMMA GEMM: C[M,N] (+)= A[M,K]@B[K,N] --------------
// B is supplied TRANSPOSED: Bt[n][k] row-major (ldbt = K typically).
// MODE 0: C f32 = acc ; MODE 1: C f32 += acc ; MODE 2: Ch f16 = lrelu(acc+bias)
#define GBM 128
#define GBN 64
#define GBK 32
#define ALD (GBK + 8)   // 40 halves -> 80B row stride (16B aligned)

template <int MODE>
__global__ void __launch_bounds__(256)
wmma_gemm_kernel(const _Float16* __restrict__ A, int lda,
                 const _Float16* __restrict__ Bt, int ldbt,
                 float* __restrict__ C, _Float16* __restrict__ Ch, int ldc,
                 const float* __restrict__ biasv, int K) {
  __shared__ _Float16 As[GBM * ALD];   // [m][k] 128x(32+8)
  __shared__ _Float16 Bs[GBN * ALD];   // [n][k]  64x(32+8)
  const int bm = blockIdx.y * GBM, bn = blockIdx.x * GBN;
  const int tid = threadIdx.x, wave = tid >> 5, lane = tid & 31;
  const int ar = tid >> 1, acb = (tid & 1) << 4;   // A: 128 rows x 32 k
  const int br = tid >> 2, bcb = (tid & 3) << 3;   // B: 64 rows x 32 k
  v8f acc[4] = {};
  for (int k0 = 0; k0 < K; k0 += GBK) {
    const float4* ap = reinterpret_cast<const float4*>(A + (size_t)(bm + ar) * lda + k0 + acb);
    const float4 a0 = ap[0], a1 = ap[1];
    const float4 b0 = *reinterpret_cast<const float4*>(Bt + (size_t)(bn + br) * ldbt + k0 + bcb);
    __syncthreads();
    *reinterpret_cast<float4*>(&As[ar * ALD + acb])     = a0;
    *reinterpret_cast<float4*>(&As[ar * ALD + acb + 8]) = a1;
    *reinterpret_cast<float4*>(&Bs[br * ALD + bcb])     = b0;
    __syncthreads();
    const v16h af = load_a_frag(&As[(wave << 4) * ALD], ALD);
#pragma unroll
    for (int nb = 0; nb < 4; ++nb) {
      const v16h bf = load_b_frag_nk(&Bs[(nb * 16) * ALD], ALD);
      acc[nb] = WMMA32(af, bf, acc[nb]);
    }
  }
  const int cc = lane & 15, half = lane >> 4;
  const int rbase = bm + (wave << 4) + (half << 3);
#pragma unroll
  for (int nb = 0; nb < 4; ++nb) {
    const int col = bn + nb * 16 + cc;
#pragma unroll
    for (int i = 0; i < 8; ++i) {
      const size_t off = (size_t)(rbase + i) * ldc + col;
      float v = acc[nb][i];
      if constexpr (MODE == 0) {
        C[off] = v;
      } else if constexpr (MODE == 1) {
        C[off] += v;
      } else {
        v += biasv[col];
        v = v > 0.f ? v : 0.1f * v;
        Ch[off] = (_Float16)v;
      }
    }
  }
}

// -------- fused GEGLU GEMM: h = gelu(A@W1g) * (A@W1v), f16 out [M,FFP_] ----
// W1pt is transposed+padded: [2*FFP_][512] (rows 0..FFP_-1 = val, FFP_.. = gate)
__global__ void __launch_bounds__(256)
ff_gemm_kernel(const _Float16* __restrict__ A, const _Float16* __restrict__ W1pt,
               _Float16* __restrict__ Hout) {
  __shared__ _Float16 As[GBM * ALD];
  __shared__ _Float16 Bv[GBN * ALD];
  __shared__ _Float16 Bg[GBN * ALD];
  const int bm = blockIdx.y * GBM, bn = blockIdx.x * GBN;
  const int tid = threadIdx.x, wave = tid >> 5, lane = tid & 31;
  const int ar = tid >> 1, acb = (tid & 1) << 4;
  const int br = tid >> 2, bcb = (tid & 3) << 3;
  v8f av[4] = {}, ag[4] = {};
  for (int k0 = 0; k0 < DIM_; k0 += GBK) {
    const float4* ap = reinterpret_cast<const float4*>(A + (size_t)(bm + ar) * DIM_ + k0 + acb);
    const float4 a0 = ap[0], a1 = ap[1];
    const float4 bv0 =
        *reinterpret_cast<const float4*>(W1pt + (size_t)(bn + br) * DIM_ + k0 + bcb);
    const float4 bg0 =
        *reinterpret_cast<const float4*>(W1pt + (size_t)(FFP_ + bn + br) * DIM_ + k0 + bcb);
    __syncthreads();
    *reinterpret_cast<float4*>(&As[ar * ALD + acb])     = a0;
    *reinterpret_cast<float4*>(&As[ar * ALD + acb + 8]) = a1;
    *reinterpret_cast<float4*>(&Bv[br * ALD + bcb])     = bv0;
    *reinterpret_cast<float4*>(&Bg[br * ALD + bcb])     = bg0;
    __syncthreads();
    const v16h af = load_a_frag(&As[(wave << 4) * ALD], ALD);
#pragma unroll
    for (int nb = 0; nb < 4; ++nb) {
      const v16h bfv = load_b_frag_nk(&Bv[(nb * 16) * ALD], ALD);
      av[nb] = WMMA32(af, bfv, av[nb]);
      const v16h bfg = load_b_frag_nk(&Bg[(nb * 16) * ALD], ALD);
      ag[nb] = WMMA32(af, bfg, ag[nb]);
    }
  }
  const int cc = lane & 15, half = lane >> 4;
  const int rbase = bm + (wave << 4) + (half << 3);
#pragma unroll
  for (int nb = 0; nb < 4; ++nb) {
    const int col = bn + nb * 16 + cc;
#pragma unroll
    for (int i = 0; i < 8; ++i) {
      const float val = av[nb][i], g = ag[nb][i];
      const float ge = 0.5f * g * (1.0f + erff(g * 0.7071067811865476f));
      Hout[(size_t)(rbase + i) * FFP_ + col] = (_Float16)(ge * val);
    }
  }
}

// ---------------- attention: one block per (bt,h), 8 waves ------------------
// qh/kh: [bh][n][d] f16 ; vht: [bh][d][n] f16 (pre-transposed by qknorm)
#define KLD 80     // 80 halves = 160B row stride (16B aligned)
#define VLD 264    // 264 halves = 528B (16B multiple)
#define PLD 272    // 272 halves = 544B (16B multiple)
__global__ void __launch_bounds__(256)
attn_kernel(const _Float16* __restrict__ qh, const _Float16* __restrict__ kh,
            const _Float16* __restrict__ vht, const float* __restrict__ bias,
            _Float16* __restrict__ o16) {
  extern __shared__ char smem[];
  _Float16* Ksh  = (_Float16*)smem;          // [n=256][d] KLD
  _Float16* Vsh  = Ksh + 256 * KLD;          // [d=64][n]  VLD (transposed)
  _Float16* Qall = Vsh + 64 * VLD;           // 8 waves x [16][d] KLD
  _Float16* Pall = Qall + 8 * 16 * KLD;      // 8 waves x [16][n] PLD
  const int bh = blockIdx.x;
  const int h  = bh & 7;
  const size_t bt = (size_t)(bh >> 3);
  const int tid = threadIdx.x, wave = tid >> 5, lane = tid & 31;
  _Float16* Qsh = Qall + wave * 16 * KLD;
  _Float16* Psh = Pall + wave * 16 * PLD;
  // cooperative K load: row per thread (128B)
  {
    const _Float16* Kg = kh + (size_t)bh * (256 * 64) + (size_t)tid * 64;
#pragma unroll
    for (int u = 0; u < 8; ++u)
      *reinterpret_cast<float4*>(&Ksh[tid * KLD + u * 8]) =
          reinterpret_cast<const float4*>(Kg)[u];
  }
  // cooperative V^T load: 64 rows x 256 cols; thread: d = t/4, 64 cols
  {
    const int vd = tid >> 2, vn = (tid & 3) << 6;
    const _Float16* Vg = vht + (size_t)bh * (64 * 256) + (size_t)vd * 256 + vn;
#pragma unroll
    for (int u = 0; u < 8; ++u)
      *reinterpret_cast<float4*>(&Vsh[vd * VLD + vn + u * 8]) =
          reinterpret_cast<const float4*>(Vg)[u];
  }
  __syncthreads();
  const int cc = lane & 15, half = lane >> 4;
  const float* brow = bias + (size_t)h * 65536;
#pragma unroll 1
  for (int t = 0; t < 2; ++t) {
    const int r0 = (wave + t * 8) << 4;   // 16 query rows for this wave
    {  // wave-local Q tile stage
      const int qr = lane >> 1, qc = (lane & 1) * 32;
      const _Float16* Qg = qh + (size_t)bh * (256 * 64) + (size_t)(r0 + qr) * 64 + qc;
#pragma unroll
      for (int u = 0; u < 4; ++u)
        *reinterpret_cast<float4*>(&Qsh[qr * KLD + qc + u * 8]) =
            reinterpret_cast<const float4*>(Qg)[u];
    }
    v8f acc[16] = {};
#pragma unroll
    for (int ks = 0; ks < 2; ++ks) {
      const v16h a = load_a_frag(&Qsh[ks * 32], KLD);
#pragma unroll
      for (int nt = 0; nt < 16; ++nt) {
        const v16h b = load_b_frag_nk(&Ksh[(nt * 16) * KLD + ks * 32], KLD);
        acc[nt] = WMMA32(a, b, acc[nt]);
      }
    }
    // softmax per row (row = r0 + 8*half + i ; cols striped over 16 lanes x 16 tiles)
#pragma unroll
    for (int i = 0; i < 8; ++i) {
      const int row = r0 + (half << 3) + i;
      float vals[16];
      float mx = -3.0e38f;
#pragma unroll
      for (int nt = 0; nt < 16; ++nt) {
        const float v = acc[nt][i] * 8.0f + brow[(size_t)row * 256 + nt * 16 + cc];
        vals[nt] = v;
        mx = fmaxf(mx, v);
      }
#pragma unroll
      for (int m2 = 8; m2; m2 >>= 1) mx = fmaxf(mx, __shfl_xor(mx, m2, 32));
      float s = 0.f;
#pragma unroll
      for (int nt = 0; nt < 16; ++nt) {
        const float e = __expf(vals[nt] - mx);
        vals[nt] = e;
        s += e;
      }
#pragma unroll
      for (int m2 = 8; m2; m2 >>= 1) s += __shfl_xor(s, m2, 32);
      const float inv = 1.0f / s;
#pragma unroll
      for (int nt = 0; nt < 16; ++nt)
        Psh[((half << 3) + i) * PLD + nt * 16 + cc] = (_Float16)(vals[nt] * inv);
    }
    // O = P @ V  (V in LDS is [d][n] -> nk fragments are contiguous)
    v8f oacc[4] = {};
#pragma unroll
    for (int jb = 0; jb < 8; ++jb) {
      const v16h a = load_a_frag(&Psh[jb * 32], PLD);
#pragma unroll
      for (int nt = 0; nt < 4; ++nt) {
        const v16h b = load_b_frag_nk(&Vsh[(nt * 16) * VLD + jb * 32], VLD);
        oacc[nt] = WMMA32(a, b, oacc[nt]);
      }
    }
#pragma unroll
    for (int nt = 0; nt < 4; ++nt)
#pragma unroll
      for (int i = 0; i < 8; ++i) {
        const int row = r0 + (half << 3) + i;
        o16[(bt * 256 + row) * 512 + h * 64 + nt * 16 + cc] = (_Float16)oacc[nt][i];
      }
  }
}

// ---------------- elementwise / reduction kernels ---------------------------
__global__ void __launch_bounds__(256)
patchify_ln_kernel(const float* __restrict__ video, const float* __restrict__ g,
                   const float* __restrict__ b, _Float16* __restrict__ xp) {
  __shared__ float rs1[8], rs2[8];
  const int m = blockIdx.x;                 // token
  const int bt = m >> 8, n = m & 255;
  const int bb = bt >> 4, tt = bt & 15;
  const int hp = n >> 4, wp = n & 15;
  const int d0 = threadIdx.x, d1 = threadIdx.x + 256;
  auto ld = [&](int d) -> float {
    const int pt = d >> 8, ph = (d >> 4) & 15, pw = d & 15;
    const size_t vi = (((size_t)(bb * 32 + tt * 2 + pt)) * 256 + (hp * 16 + ph)) * 256 +
                      (wp * 16 + pw);
    return video[vi];
  };
  const float v0 = ld(d0), v1 = ld(d1);
  float s = v0 + v1, s2 = v0 * v0 + v1 * v1;
  const int lane = threadIdx.x & 31, wv = threadIdx.x >> 5;
#pragma unroll
  for (int o = 16; o; o >>= 1) { s += __shfl_xor(s, o, 32); s2 += __shfl_xor(s2, o, 32); }
  if (lane == 0) { rs1[wv] = s; rs2[wv] = s2; }
  __syncthreads();
  float S = 0.f, S2 = 0.f;
#pragma unroll
  for (int u = 0; u < 8; ++u) { S += rs1[u]; S2 += rs2[u]; }
  const float mu = S * (1.f / 512.f);
  const float var = S2 * (1.f / 512.f) - mu * mu;
  const float rstd = rsqrtf(var + 1e-5f);
  xp[(size_t)m * 512 + d0] = (_Float16)((v0 - mu) * rstd * g[d0] + b[d0]);
  xp[(size_t)m * 512 + d1] = (_Float16)((v1 - mu) * rstd * g[d1] + b[d1]);
}

template <bool F16OUT>
__global__ void __launch_bounds__(256)
ln_kernel(const float* __restrict__ x, const float* __restrict__ g,
          const float* __restrict__ b, float* __restrict__ of,
          _Float16* __restrict__ oh) {
  __shared__ float rs1[8], rs2[8];
  const int m = blockIdx.x;
  const int d0 = threadIdx.x, d1 = threadIdx.x + 256;
  const float v0 = x[(size_t)m * 512 + d0], v1 = x[(size_t)m * 512 + d1];
  float s = v0 + v1, s2 = v0 * v0 + v1 * v1;
  const int lane = threadIdx.x & 31, wv = threadIdx.x >> 5;
#pragma unroll
  for (int o = 16; o; o >>= 1) { s += __shfl_xor(s, o, 32); s2 += __shfl_xor(s2, o, 32); }
  if (lane == 0) { rs1[wv] = s; rs2[wv] = s2; }
  __syncthreads();
  float S = 0.f, S2 = 0.f;
#pragma unroll
  for (int u = 0; u < 8; ++u) { S += rs1[u]; S2 += rs2[u]; }
  const float mu = S * (1.f / 512.f);
  const float rstd = rsqrtf(S2 * (1.f / 512.f) - mu * mu + 1e-5f);
  const float b0 = b ? b[d0] : 0.f, b1 = b ? b[d1] : 0.f;
  const float y0 = (v0 - mu) * rstd * g[d0] + b0;
  const float y1 = (v1 - mu) * rstd * g[d1] + b1;
  if constexpr (F16OUT) {
    oh[(size_t)m * 512 + d0] = (_Float16)y0;
    oh[(size_t)m * 512 + d1] = (_Float16)y1;
  } else {
    of[(size_t)m * 512 + d0] = y0;
    of[(size_t)m * 512 + d1] = y1;
  }
}

// q/k: l2norm * scale -> [bh][n][d] f16 ; v: convert -> [bh][d][n] f16 (transposed)
__global__ void __launch_bounds__(256)
qknorm_kernel(const float* __restrict__ q, const float* __restrict__ kv,
              const float* __restrict__ qs, const float* __restrict__ ks2,
              _Float16* __restrict__ qh, _Float16* __restrict__ kh,
              _Float16* __restrict__ vht) {
  const int m = blockIdx.x;
  const int wv = threadIdx.x >> 5, lane = threadIdx.x & 31;  // wave = head
  const int bt = m >> 8, n = m & 255;
  const float* qr = q  + (size_t)m * 512  + wv * 64;
  const float* kr = kv + (size_t)m * 1024 + wv * 64;
  const float* vr = kv + (size_t)m * 1024 + 512 + wv * 64;
  const float q0 = qr[lane], q1 = qr[lane + 32];
  const float k0 = kr[lane], k1 = kr[lane + 32];
  float sq = q0 * q0 + q1 * q1, sk = k0 * k0 + k1 * k1;
#pragma unroll
  for (int o = 16; o; o >>= 1) { sq += __shfl_xor(sq, o, 32); sk += __shfl_xor(sk, o, 32); }
  const float rq = 1.f / fmaxf(sqrtf(sq), 1e-12f);
  const float rk = 1.f / fmaxf(sqrtf(sk), 1e-12f);
  const size_t bh = (size_t)bt * 8 + wv;
  const size_t base = (bh * 256 + n) * 64;
  qh[base + lane]      = (_Float16)(q0 * rq * qs[lane]);
  qh[base + lane + 32] = (_Float16)(q1 * rq * qs[lane + 32]);
  kh[base + lane]      = (_Float16)(k0 * rk * ks2[lane]);
  kh[base + lane + 32] = (_Float16)(k1 * rk * ks2[lane + 32]);
  const size_t vb = bh * (64 * 256) + n;           // [bh][d][n]
  vht[vb + (size_t)lane * 256]        = (_Float16)vr[lane];
  vht[vb + (size_t)(lane + 32) * 256] = (_Float16)vr[lane + 32];
}

__global__ void __launch_bounds__(256)
peg_conv_kernel(const float* __restrict__ x, const float* __restrict__ w,
                const float* __restrict__ bvec, float* __restrict__ y) {
  const size_t idx = (size_t)blockIdx.x * 256 + threadIdx.x;  // MTOK_*512
  const int c = (int)(idx & 511);
  const size_t m = idx >> 9;
  const int n = (int)(m & 255), bt = (int)(m >> 8);
  const int wp = n & 15, hp = n >> 4;
  const int tt = bt & 15, bb = bt >> 4;
  float s = bvec[c];
#pragma unroll
  for (int dt = 0; dt < 3; ++dt) {
    const int t2 = tt + dt - 2;
    if (t2 < 0 || t2 >= TT_) continue;
#pragma unroll
    for (int dh = 0; dh < 3; ++dh) {
      const int h2 = hp + dh - 1;
      if (h2 < 0 || h2 >= 16) continue;
#pragma unroll
      for (int dw = 0; dw < 3; ++dw) {
        const int w2 = wp + dw - 1;
        if (w2 < 0 || w2 >= 16) continue;
        const size_t xi = (((size_t)(bb * 16 + t2) * 256) + h2 * 16 + w2) * 512 + c;
        s += w[c * 27 + dt * 9 + dh * 3 + dw] * x[xi];
      }
    }
  }
  y[idx] = s;
}

__global__ void axpy_kernel(float* __restrict__ x, const float* __restrict__ y, size_t n) {
  const size_t i = (size_t)blockIdx.x * 256 + threadIdx.x;
  if (i < n) x[i] += y[i];
}

__device__ __forceinline__ float siglog(float r) {
  const float a = logf(fabsf(r) + 1.0f);
  return (r > 0.f) ? a : ((r < 0.f) ? -a : 0.0f);
}

__global__ void cpb_z0_kernel(const float* __restrict__ w0, const float* __restrict__ b0,
                              _Float16* __restrict__ z0) {
  const size_t idx = (size_t)blockIdx.x * 256 + threadIdx.x;  // NPAIR_*512
  const int oc = (int)(idx & 511);
  const size_t p = idx >> 9;
  const int i = (int)(p >> 8), j = (int)(p & 255);
  const float r0 = siglog((float)(i >> 4) - (float)(j >> 4));
  const float r1 = siglog((float)(i & 15) - (float)(j & 15));
  const float v = r0 * w0[oc] + r1 * w0[512 + oc] + b0[oc];
  z0[idx] = (_Float16)(v > 0.f ? v : 0.1f * v);
}

__global__ void cpb_out_kernel(const _Float16* __restrict__ z1, const float* __restrict__ w2,
                               const float* __restrict__ b2, float* __restrict__ bias) {
  const size_t idx = (size_t)blockIdx.x * 256 + threadIdx.x;  // NPAIR_*8
  const int h = (int)(idx & 7);
  const size_t p = idx >> 3;
  const _Float16* zr = z1 + p * 512;
  float s = b2[h];
  for (int k = 0; k < 512; ++k) s += (float)zr[k] * w2[k * 8 + h];
  bias[(size_t)h * 65536 + p] = s;
}

// ---------------- weight conversion (all TRANSPOSED to [n][k] f16) ----------
// src: L matrices [K][N] f32 ; dst: L matrices [N][K] f16
__global__ void f32_to_f16_t_kernel(const float* __restrict__ s, _Float16* __restrict__ d,
                                    int K, int N, size_t total) {
  const size_t idx = (size_t)blockIdx.x * 256 + threadIdx.x;
  if (idx >= total) return;
  const size_t mat = (size_t)K * N;
  const size_t l = idx / mat, rem = idx - l * mat;
  const int n = (int)(rem / K), k = (int)(rem - (size_t)n * K);
  d[idx] = (_Float16)s[l * mat + (size_t)k * N + n];
}
// ff_w1 [D][512][2730] -> transposed+padded [D][2*1408][512] f16
__global__ void pad_w1t_kernel(const float* __restrict__ s, _Float16* __restrict__ d) {
  const size_t idx = (size_t)blockIdx.x * 256 + threadIdx.x;  // D*2816*512
  const int k = (int)(idx & 511);
  const size_t lj = idx >> 9;
  const size_t l = lj / (2 * FFP_);
  const int j = (int)(lj - l * (2 * FFP_));
  const int half = j >= FFP_;
  const int jj = j - half * FFP_;
  _Float16 v = (_Float16)0.f;
  if (jj < FFI_)
    v = (_Float16)s[(l * 512 + k) * (2 * FFI_) + half * FFI_ + jj];
  d[idx] = v;
}
// ff_w2 [D][1365][512] -> transposed+padded [D][512][1408] f16
__global__ void pad_w2t_kernel(const float* __restrict__ s, _Float16* __restrict__ d) {
  const size_t idx = (size_t)blockIdx.x * 256 + threadIdx.x;  // D*512*1408
  const int k = (int)(idx % FFP_);
  const size_t ln = idx / FFP_;
  const size_t l = ln >> 9;
  const int n = (int)(ln & 511);
  _Float16 v = (_Float16)0.f;
  if (k < FFI_) v = (_Float16)s[(l * FFI_ + k) * 512 + n];
  d[idx] = v;
}

// ---------------- host launcher ----------------
extern "C" void kernel_launch(void* const* d_in, const int* in_sizes, int n_in,
                              void* d_out, int out_size, void* d_ws, size_t ws_size,
                              hipStream_t stream) {
  (void)in_sizes; (void)n_in; (void)out_size; (void)ws_size;
  const float* video    = (const float*)d_in[0];
  const float* p_ln1_g  = (const float*)d_in[1];
  const float* p_ln1_b  = (const float*)d_in[2];
  const float* patch_w  = (const float*)d_in[3];
  const float* p_ln2_g  = (const float*)d_in[4];
  const float* p_ln2_b  = (const float*)d_in[5];
  const float* cpb_w0   = (const float*)d_in[6];
  const float* cpb_b0   = (const float*)d_in[7];
  const float* cpb_w1   = (const float*)d_in[8];
  const float* cpb_b1   = (const float*)d_in[9];
  const float* cpb_w2   = (const float*)d_in[10];
  const float* cpb_b2   = (const float*)d_in[11];
  const float* peg_w    = (const float*)d_in[12];
  const float* peg_b    = (const float*)d_in[13];
  const float* attn_g   = (const float*)d_in[14];
  const float* wq       = (const float*)d_in[15];
  const float* wkv      = (const float*)d_in[16];
  const float* q_scale  = (const float*)d_in[17];
  const float* k_scale  = (const float*)d_in[18];
  const float* wo       = (const float*)d_in[19];
  const float* ff_g     = (const float*)d_in[20];
  const float* ff_b     = (const float*)d_in[21];
  const float* ff_w1    = (const float*)d_in[22];
  const float* ff_w2    = (const float*)d_in[23];
  const float* out_g    = (const float*)d_in[24];

  char* ws = (char*)d_ws;
  size_t off = 0;
  auto alloc = [&](size_t bytes) { size_t o = off; off += (bytes + 255) & ~(size_t)255; return o; };
  const size_t off_x   = alloc((size_t)MTOK_ * 512 * 4);           // residual stream f32
  const size_t off_bias= alloc((size_t)8 * 65536 * 4);             // CPB bias f32
  const size_t off_pw  = alloc((size_t)512 * 512 * 2);             // f16 weights (transposed)
  const size_t off_cw1 = alloc((size_t)512 * 512 * 2);
  const size_t off_wq  = alloc((size_t)DEPTH_ * 512 * 512 * 2);
  const size_t off_wkv = alloc((size_t)DEPTH_ * 512 * 1024 * 2);
  const size_t off_wo  = alloc((size_t)DEPTH_ * 512 * 512 * 2);
  const size_t off_w1p = alloc((size_t)DEPTH_ * 512 * 2 * FFP_ * 2);
  const size_t off_w2p = alloc((size_t)DEPTH_ * FFP_ * 512 * 2);
  const size_t off_xn  = alloc((size_t)MTOK_ * 512 * 2);           // f16 LN'd activations
  const size_t off_q   = alloc((size_t)MTOK_ * 512 * 4);           // q f32 (alias: ptmp, z0)
  const size_t off_kv  = alloc((size_t)MTOK_ * 1024 * 4);          // kv f32 (z0 spills in here)
  const size_t off_qh  = alloc((size_t)MTOK_ * 512 * 2);           // head-major f16 (alias: z1)
  const size_t off_kh  = alloc((size_t)MTOK_ * 512 * 2);
  const size_t off_vh  = alloc((size_t)MTOK_ * 512 * 2);
  const size_t off_o16 = alloc((size_t)MTOK_ * 512 * 2);
  const size_t off_h   = alloc((size_t)MTOK_ * FFP_ * 2);

  float*     x     = (float*)(ws + off_x);
  float*     biasb = (float*)(ws + off_bias);
  _Float16*  pw16  = (_Float16*)(ws + off_pw);
  _Float16*  cw116 = (_Float16*)(ws + off_cw1);
  _Float16*  wq16  = (_Float16*)(ws + off_wq);
  _Float16*  wkv16 = (_Float16*)(ws + off_wkv);
  _Float16*  wo16  = (_Float16*)(ws + off_wo);
  _Float16*  w1p16 = (_Float16*)(ws + off_w1p);
  _Float16*  w2p16 = (_Float16*)(ws + off_w2p);
  _Float16*  xn16  = (_Float16*)(ws + off_xn);
  float*     qbuf  = (float*)(ws + off_q);
  float*     kvbuf = (float*)(ws + off_kv);
  float*     ptmp  = (float*)(ws + off_q);      // alias (q dead when used)
  _Float16*  z0    = (_Float16*)(ws + off_q);   // alias: 64MB spans q+kv
  _Float16*  z1    = (_Float16*)(ws + off_qh);  // alias: 64MB spans qh..o16
  _Float16*  qhb   = (_Float16*)(ws + off_qh);
  _Float16*  khb   = (_Float16*)(ws + off_kh);
  _Float16*  vhb   = (_Float16*)(ws + off_vh);
  _Float16*  o16   = (_Float16*)(ws + off_o16);
  _Float16*  hbuf  = (_Float16*)(ws + off_h);

  const dim3 blk(256);
  auto g1 = [](size_t n) { return dim3((unsigned)((n + 255) / 256)); };

  // ---- weight conversion: f32 [K][N] -> f16 [N][K] (transposed) ----
  f32_to_f16_t_kernel<<<g1((size_t)512 * 512), blk, 0, stream>>>(
      patch_w, pw16, 512, 512, (size_t)512 * 512);
  f32_to_f16_t_kernel<<<g1((size_t)512 * 512), blk, 0, stream>>>(
      cpb_w1, cw116, 512, 512, (size_t)512 * 512);
  f32_to_f16_t_kernel<<<g1((size_t)DEPTH_ * 512 * 512), blk, 0, stream>>>(
      wq, wq16, 512, 512, (size_t)DEPTH_ * 512 * 512);
  f32_to_f16_t_kernel<<<g1((size_t)DEPTH_ * 512 * 1024), blk, 0, stream>>>(
      wkv, wkv16, 512, 1024, (size_t)DEPTH_ * 512 * 1024);
  f32_to_f16_t_kernel<<<g1((size_t)DEPTH_ * 512 * 512), blk, 0, stream>>>(
      wo, wo16, 512, 512, (size_t)DEPTH_ * 512 * 512);
  pad_w1t_kernel<<<g1((size_t)DEPTH_ * 2 * FFP_ * 512), blk, 0, stream>>>(ff_w1, w1p16);
  pad_w2t_kernel<<<g1((size_t)DEPTH_ * 512 * FFP_), blk, 0, stream>>>(ff_w2, w2p16);

  // ---- patch embed: gather+LN1 -> f16, GEMM, LN2 -> x ----
  patchify_ln_kernel<<<MTOK_, blk, 0, stream>>>(video, p_ln1_g, p_ln1_b, xn16);
  wmma_gemm_kernel<0><<<dim3(512 / GBN, MTOK_ / GBM), blk, 0, stream>>>(
      xn16, 512, pw16, 512, ptmp, nullptr, 512, nullptr, 512);
  ln_kernel<false><<<MTOK_, blk, 0, stream>>>(ptmp, p_ln2_g, p_ln2_b, x, nullptr);

  // ---- CPB relative position bias ----
  cpb_z0_kernel<<<g1((size_t)NPAIR_ * 512), blk, 0, stream>>>(cpb_w0, cpb_b0, z0);
  wmma_gemm_kernel<2><<<dim3(512 / GBN, NPAIR_ / GBM), blk, 0, stream>>>(
      z0, 512, cw116, 512, nullptr, z1, 512, cpb_b1, 512);
  cpb_out_kernel<<<g1((size_t)NPAIR_ * 8), blk, 0, stream>>>(z1, cpb_w2, cpb_b2, biasb);

  const size_t attn_lds =
      (size_t)(256 * KLD + 64 * VLD + 8 * 16 * KLD + 8 * 16 * PLD) * 2;

  // ---- transformer layers ----
  for (int l = 0; l < DEPTH_; ++l) {
    // PEG depthwise causal conv + residual
    peg_conv_kernel<<<g1((size_t)MTOK_ * 512), blk, 0, stream>>>(
        x, peg_w + (size_t)l * 512 * 27, peg_b + (size_t)l * 512, ptmp);
    axpy_kernel<<<g1((size_t)MTOK_ * 512), blk, 0, stream>>>(x, ptmp, (size_t)MTOK_ * 512);
    // attention
    ln_kernel<true><<<MTOK_, blk, 0, stream>>>(x, attn_g + (size_t)l * 512, nullptr, nullptr, xn16);
    wmma_gemm_kernel<0><<<dim3(512 / GBN, MTOK_ / GBM), blk, 0, stream>>>(
        xn16, 512, wq16 + (size_t)l * 512 * 512, 512, qbuf, nullptr, 512, nullptr, 512);
    wmma_gemm_kernel<0><<<dim3(1024 / GBN, MTOK_ / GBM), blk, 0, stream>>>(
        xn16, 512, wkv16 + (size_t)l * 512 * 1024, 512, kvbuf, nullptr, 1024, nullptr, 512);
    qknorm_kernel<<<MTOK_, blk, 0, stream>>>(qbuf, kvbuf, q_scale + (size_t)l * 64,
                                             k_scale + (size_t)l * 64, qhb, khb, vhb);
    attn_kernel<<<BT_ * HEADS_, blk, attn_lds, stream>>>(qhb, khb, vhb, biasb, o16);
    wmma_gemm_kernel<1><<<dim3(512 / GBN, MTOK_ / GBM), blk, 0, stream>>>(
        o16, 512, wo16 + (size_t)l * 512 * 512, 512, x, nullptr, 512, nullptr, 512);
    // GEGLU FF
    ln_kernel<true><<<MTOK_, blk, 0, stream>>>(x, ff_g + (size_t)l * 512, ff_b + (size_t)l * 512,
                                               nullptr, xn16);
    ff_gemm_kernel<<<dim3(FFP_ / GBN, MTOK_ / GBM), blk, 0, stream>>>(
        xn16, w1p16 + (size_t)l * 512 * 2 * FFP_, hbuf);
    wmma_gemm_kernel<1><<<dim3(512 / GBN, MTOK_ / GBM), blk, 0, stream>>>(
        hbuf, FFP_, w2p16 + (size_t)l * FFP_ * 512, 512, x, nullptr, 512, nullptr, FFP_);
  }

  // ---- final LN -> output ----
  ln_kernel<false><<<MTOK_, blk, 0, stream>>>(x, out_g, nullptr, (float*)d_out, nullptr);
}
